// ModAFNO2DLayer_67233418051735
// MI455X (gfx1250) — matrix operand order repaired and checked by reading, not compiled, not run on hardware
//
#include <hip/hip_runtime.h>
#include <hip/hip_bf16.h>
#include <stdint.h>

#define DIMC 768
#define NB   8
#define BS   96
#define BSZ  4
#define HH   128
#define WW   128
#define WF   65
#define LAM  0.01f

static constexpr size_t NPOS   = (size_t)BSZ * HH * WF;      // 33280 frequency positions
static constexpr size_t NTILES = NPOS / 16;                  // 2080
static constexpr size_t NCPLX  = (size_t)BSZ * DIMC * HH * WF;
static constexpr float  ORTHO  = 0.08838834764831845f;       // 1/sqrt(128)

typedef __attribute__((ext_vector_type(16))) __bf16 v16bf;
typedef __attribute__((ext_vector_type(8)))  float  v8f;

union Frag { v16bf v; uint32_t u[8]; uint4 q[2]; };

// ---------------- scratch layout (bytes) ----------------
static constexpr size_t MOD_OFF  = 0;
static constexpr size_t WBF_OFF  = 24576;                              // B*2*DIM*4
static constexpr size_t SPECA_OFF = WBF_OFF + (size_t)NB*4*BS*BS*2;    // bf16 weights
static constexpr size_t SBR_OFF  = SPECA_OFF + NCPLX * 8;              // f32 complex spectra
static constexpr size_t SBI_OFF  = SBR_OFF + NCPLX * 2;                // bf16 planes

// ================= modulation: silu(t) @ mod_w^T + mod_b =================
__global__ void k_mod(const float* __restrict__ t, const float* __restrict__ mw,
                      const float* __restrict__ mb, float* __restrict__ modbuf) {
    __shared__ float ts[DIMC];
    const int tid = threadIdx.x;
    const int jg  = blockIdx.x * 256 + tid;          // 0 .. B*1536-1
    const int b   = jg / (2 * DIMC);
    const int j   = jg % (2 * DIMC);
    for (int k = tid; k < DIMC; k += 256) {
        float v = t[b * DIMC + k];
        ts[k] = v / (1.0f + __expf(-v));
    }
    __syncthreads();
    float acc = mb[j];
    const float* wrow = mw + (size_t)j * DIMC;
    for (int k = 0; k < DIMC; ++k) acc = fmaf(ts[k], wrow[k], acc);
    modbuf[b * 2 * DIMC + j] = acc;
}

// ====== weights -> bf16, transposed to [nb][mat][n][d]  (mat: w1r,w1i,w2r,w2i) ======
__global__ void k_wconv(const float* __restrict__ w1, const float* __restrict__ w2,
                        __bf16* __restrict__ wbf) {
    int idx = blockIdx.x * 256 + threadIdx.x;        // NB*4*96*96 = 294912
    if (idx >= NB * 4 * BS * BS) return;
    int d   = idx % BS;
    int n   = (idx / BS) % BS;
    int mat = (idx / (BS * BS)) % 4;
    int nb  = idx / (4 * BS * BS);
    const float* src = (mat < 2) ? w1 : w2;
    int ri = mat & 1;
    float v = src[(((size_t)ri * NB + nb) * BS + d) * BS + n];
    wbf[idx] = (__bf16)v;
}

// ================= shared 128-point radix-2 FFT in LDS =================
__device__ __forceinline__ void fft128(float* re, float* im, float sign) {
    const int t = threadIdx.x;
    int j = __brev((unsigned)t) >> 25;               // 7-bit reverse
    if (j > t) {
        float a = re[t], b = im[t];
        re[t] = re[j]; im[t] = im[j];
        re[j] = a;     im[j] = b;
    }
    __syncthreads();
    for (int len = 2; len <= 128; len <<= 1) {
        int half = len >> 1;
        if (t < 64) {
            int blk = t / half, pos = t % half;
            int i0 = blk * len + pos, i1 = i0 + half;
            float s, c;
            __sincosf(sign * 6.283185307179586f * (float)pos / (float)len, &s, &c);
            float xr = re[i1], xi = im[i1];
            float vr = fmaf(xr, c, -xi * s);
            float vi = fmaf(xr, s,  xi * c);
            float ur = re[i0], ui = im[i0];
            re[i1] = ur - vr; im[i1] = ui - vi;
            re[i0] = ur + vr; im[i0] = ui + vi;
        }
        __syncthreads();
    }
}

// ======== forward rfft along W: x row -> specA[b][c][h][wf] (f32 complex) ========
__global__ void k_rowfft(const float* __restrict__ x, float2* __restrict__ specA) {
    __shared__ float re[128], im[128];
    const size_t r = blockIdx.x;                     // (b*DIM+c)*H + h
    const int t = threadIdx.x;
    re[t] = x[r * WW + t];
    im[t] = 0.0f;
    __syncthreads();
    fft128(re, im, -1.0f);
    if (t < WF) specA[r * WF + t] = make_float2(re[t] * ORTHO, im[t] * ORTHO);
}

// ======== forward FFT along H: specA -> bf16 planes [b][h][wf][c] ========
__global__ void k_colfft(const float2* __restrict__ specA,
                         __bf16* __restrict__ br, __bf16* __restrict__ bi) {
    __shared__ float re[128], im[128];
    const size_t col = blockIdx.x;                   // (b*DIM+c)*WF + wf
    const int wf = (int)(col % WF);
    const size_t bc = col / WF;
    const int t = threadIdx.x;
    float2 v = specA[(bc * HH + t) * WF + wf];
    re[t] = v.x; im[t] = v.y;
    __syncthreads();
    fft128(re, im, -1.0f);
    const int b = (int)(bc / DIMC), c = (int)(bc % DIMC);
    size_t o = (((size_t)b * HH + t) * WF + wf) * DIMC + c;
    br[o] = (__bf16)(re[t] * ORTHO);
    bi[o] = (__bf16)(im[t] * ORTHO);
}

// ================= spectral block MLP (WMMA core) =================
__device__ __forceinline__ Frag neg_frag(Frag f) {
    Frag r;
#pragma unroll
    for (int i = 0; i < 8; ++i) r.u[i] = f.u[i] ^ 0x80008000u;
    return r;
}

__global__ void __launch_bounds__(128, 4)
k_mlp(const __bf16* __restrict__ br, const __bf16* __restrict__ bi,
      const __bf16* __restrict__ wbf, const float* __restrict__ b1,
      const float* __restrict__ b2, const float* __restrict__ modbuf,
      float* __restrict__ specC) {                   // interleaved {re,im} pairs
    __shared__ __align__(16) __bf16 wlds[4 * 96 * 112];       // [mat][n][d] padded
    __shared__ __align__(16) __bf16 stg[4][2][16 * 112];      // per-wave staging r/i
    __shared__ float biasLds[4][BS];
    __shared__ float modLds[2][BS];

    const int tid  = threadIdx.x;
    const int lane = tid & 31;
    const int wv   = tid >> 5;
    const int nb   = blockIdx.y;
    const size_t tile = (size_t)blockIdx.x * 4 + wv;
    const size_t p0   = tile * 16;
    const int b = (int)((size_t)blockIdx.x * 64 / (HH * WF));  // uniform per block

    // ---- async copy of this block's 4 bf16 weight matrices into LDS ----
    const __bf16* wsrc = wbf + (size_t)nb * 4 * BS * BS;
    const uint32_t lbase = (uint32_t)(uintptr_t)&wlds[0];
    for (int idx = tid; idx < 4 * 96 * 12; idx += 128) {       // 16B chunks
        int row = idx / 12, ch = idx % 12;
        uint32_t goff = (uint32_t)(row * 192 + ch * 16);
        uint32_t laddr = lbase + (uint32_t)(row * 224 + ch * 16);
        asm volatile("global_load_async_to_lds_b128 %0, %1, %2"
                     :: "v"(laddr), "v"(goff), "s"(wsrc) : "memory");
    }
    if (tid < BS) {
        biasLds[0][tid] = b1[(0 * NB + nb) * BS + tid];
        biasLds[1][tid] = b1[(1 * NB + nb) * BS + tid];
        biasLds[2][tid] = b2[(0 * NB + nb) * BS + tid];
        biasLds[3][tid] = b2[(1 * NB + nb) * BS + tid];
        modLds[0][tid] = modbuf[b * 2 * DIMC + nb * 2 * BS + tid];        // shift
        modLds[1][tid] = modbuf[b * 2 * DIMC + nb * 2 * BS + BS + tid];   // scale
    }
    asm volatile("s_wait_asynccnt 0x0" ::: "memory");
    __syncthreads();

    const int m     = lane & 15;
    const int hi    = lane >> 4;
    const int khalf = hi * 8;
    const int cbl   = lane & 15;                    // lane column within N-tile

    // ---- A fragments: Xr, Xi (16 positions x 96 channels) ----
    const __bf16* xr0 = br + p0 * DIMC + nb * BS;
    const __bf16* xi0 = bi + p0 * DIMC + nb * BS;
    __builtin_prefetch(xr0 + (size_t)m * DIMC, 0, 0);
    __builtin_prefetch(xi0 + (size_t)m * DIMC, 0, 0);
    Frag fxr[3], fxi[3];
#pragma unroll
    for (int kc = 0; kc < 3; ++kc) {
        const __bf16* pr = xr0 + (size_t)m * DIMC + kc * 32 + khalf;
        const __bf16* pi = xi0 + (size_t)m * DIMC + kc * 32 + khalf;
        fxr[kc].q[0] = *(const uint4*)pr;  fxr[kc].q[1] = *(const uint4*)(pr + 16);
        fxi[kc].q[0] = *(const uint4*)pi;  fxi[kc].q[1] = *(const uint4*)(pi + 16);
    }

    auto loadB = [&](int mat, int kc, int nt) -> Frag {
        Frag f;
        int n  = nt * 16 + cbl;
        int kb = kc * 32 + hi * 8;
        const __bf16* p = &wlds[(mat * 96 + n) * 112 + kb];
        f.q[0] = *(const uint4*)p;
        f.q[1] = *(const uint4*)(p + 16);
        return f;
    };

    // scattered output address (element index into complex array) for row j
    auto outIdx = [&](int j, int cb) -> size_t {
        int row = j + hi * 8;
        size_t p = p0 + (size_t)row;
        int bb  = (int)(p / (HH * WF));
        int rem = (int)(p % (HH * WF));
        int h = rem / WF, wf = rem % WF;
        int c = nb * BS + cb;
        return (((size_t)bb * DIMC + c) * HH + h) * WF + wf;
    };

    __bf16* stR = &stg[wv][0][0];
    __bf16* stI = &stg[wv][1][0];

    // ---- layer 1:  r1 = Xr*W1r - Xi*W1i + b1r ;  i1 = Xr*W1i + Xi*W1r + b1i ----
#pragma unroll 2
    for (int nt = 0; nt < 6; ++nt) {
        const int cb = nt * 16 + cbl;
        const float bR = biasLds[0][cb], bI = biasLds[1][cb];
        v8f aR, aI;
#pragma unroll
        for (int j = 0; j < 8; ++j) { aR[j] = bR; aI[j] = bI; }
#pragma unroll
        for (int kc = 0; kc < 3; ++kc) {
            Frag bw1r = loadB(0, kc, nt), bw1i = loadB(1, kc, nt);
            Frag fxin = neg_frag(fxi[kc]);
            aR = __builtin_amdgcn_wmma_f32_16x16x32_bf16(false, fxr[kc].v, false, bw1r.v,
                                                         (short)0, aR, false, false);
            aR = __builtin_amdgcn_wmma_f32_16x16x32_bf16(false, fxin.v,   false, bw1i.v,
                                                         (short)0, aR, false, false);
            aI = __builtin_amdgcn_wmma_f32_16x16x32_bf16(false, fxr[kc].v, false, bw1i.v,
                                                         (short)0, aI, false, false);
            aI = __builtin_amdgcn_wmma_f32_16x16x32_bf16(false, fxi[kc].v, false, bw1r.v,
                                                         (short)0, aI, false, false);
        }
        const float sh = modLds[0][cb] + 1.0f, sc = modLds[1][cb];
#pragma unroll
        for (int j = 0; j < 8; ++j) {
            int row = j + hi * 8;
            float r = fmaf(aR[j], sh, sc); r = r > 0.0f ? r : 0.0f;
            float i = fmaf(aI[j], sh, sc); i = i > 0.0f ? i : 0.0f;
            stR[row * 112 + cb] = (__bf16)r;
            stI[row * 112 + cb] = (__bf16)i;
        }
    }
    __syncthreads();

    // ---- reload r1/i1 as A fragments ----
    Frag fr1[3], fi1[3];
#pragma unroll
    for (int kc = 0; kc < 3; ++kc) {
        const __bf16* pr = &stR[m * 112 + kc * 32 + khalf];
        const __bf16* pi = &stI[m * 112 + kc * 32 + khalf];
        fr1[kc].q[0] = *(const uint4*)pr;  fr1[kc].q[1] = *(const uint4*)(pr + 16);
        fi1[kc].q[0] = *(const uint4*)pi;  fi1[kc].q[1] = *(const uint4*)(pi + 16);
    }

    // ---- layer 2a: r2 = r1*W2r - i1*W2i + b2r ----
    // Stage linear r2 for layer 2b, then immediately softshrink + store the REAL
    // component; accumulators die per-iteration (keeps wave under 256 VGPRs).
#pragma unroll 2
    for (int nt = 0; nt < 6; ++nt) {
        const int cb = nt * 16 + cbl;
        const float bR = biasLds[2][cb];
        v8f a;
#pragma unroll
        for (int j = 0; j < 8; ++j) a[j] = bR;
#pragma unroll
        for (int kc = 0; kc < 3; ++kc) {
            Frag bw2r = loadB(2, kc, nt), bw2i = loadB(3, kc, nt);
            Frag fin = neg_frag(fi1[kc]);
            a = __builtin_amdgcn_wmma_f32_16x16x32_bf16(false, fr1[kc].v, false, bw2r.v,
                                                        (short)0, a, false, false);
            a = __builtin_amdgcn_wmma_f32_16x16x32_bf16(false, fin.v,     false, bw2i.v,
                                                        (short)0, a, false, false);
        }
#pragma unroll
        for (int j = 0; j < 8; ++j) {
            int row = j + hi * 8;
            stR[row * 112 + cb] = (__bf16)a[j];               // linear r2 for i2
            float rv = a[j];
            float rm = fabsf(rv) - LAM; rm = rm > 0.0f ? rm : 0.0f;
            specC[outIdx(j, cb) * 2 + 0] = copysignf(rm, rv); // shrunk real part
        }
    }
    __syncthreads();

    Frag fr2[3];
#pragma unroll
    for (int kc = 0; kc < 3; ++kc) {
        const __bf16* pr = &stR[m * 112 + kc * 32 + khalf];
        fr2[kc].q[0] = *(const uint4*)pr;  fr2[kc].q[1] = *(const uint4*)(pr + 16);
    }

    // ---- layer 2b: i2 = r2*W2i + i1*W2r + b2i  (reference uses linear r2) ----
#pragma unroll 2
    for (int nt = 0; nt < 6; ++nt) {
        const int cb = nt * 16 + cbl;
        const float bI = biasLds[3][cb];
        v8f a;
#pragma unroll
        for (int j = 0; j < 8; ++j) a[j] = bI;
#pragma unroll
        for (int kc = 0; kc < 3; ++kc) {
            Frag bw2r = loadB(2, kc, nt), bw2i = loadB(3, kc, nt);
            a = __builtin_amdgcn_wmma_f32_16x16x32_bf16(false, fr2[kc].v, false, bw2i.v,
                                                        (short)0, a, false, false);
            a = __builtin_amdgcn_wmma_f32_16x16x32_bf16(false, fi1[kc].v, false, bw2r.v,
                                                        (short)0, a, false, false);
        }
#pragma unroll
        for (int j = 0; j < 8; ++j) {
            float iv = a[j];
            float im2 = fabsf(iv) - LAM; im2 = im2 > 0.0f ? im2 : 0.0f;
            specC[outIdx(j, cb) * 2 + 1] = copysignf(im2, iv); // shrunk imag part
        }
    }
}

// ======== inverse FFT along H, in place on specC [b][c][h][wf] ========
__global__ void k_icolfft(float2* __restrict__ specC) {
    __shared__ float re[128], im[128];
    const size_t col = blockIdx.x;
    const int wf = (int)(col % WF);
    const size_t bc = col / WF;
    const int t = threadIdx.x;
    float2 v = specC[(bc * HH + t) * WF + wf];
    re[t] = v.x; im[t] = v.y;
    __syncthreads();
    fft128(re, im, 1.0f);
    specC[(bc * HH + t) * WF + wf] = make_float2(re[t] * ORTHO, im[t] * ORTHO);
}

// ======== inverse rfft along W (Hermitian) + residual add ========
__global__ void k_irowfft(const float2* __restrict__ specC,
                          const float* __restrict__ x, float* __restrict__ out) {
    __shared__ float re[128], im[128];
    const size_t r = blockIdx.x;                    // (b*DIM+c)*H + h
    const int t = threadIdx.x;
    if (t < WF) {
        float2 v = specC[r * WF + t];
        re[t] = v.x; im[t] = v.y;
    }
    __syncthreads();
    if (t >= WF) { re[t] = re[128 - t]; im[t] = -im[128 - t]; }
    __syncthreads();
    fft128(re, im, 1.0f);
    out[r * WW + t] = fmaf(re[t], ORTHO, x[r * WW + t]);
}

// ================= launcher =================
extern "C" void kernel_launch(void* const* d_in, const int* in_sizes, int n_in,
                              void* d_out, int out_size, void* d_ws, size_t ws_size,
                              hipStream_t stream) {
    const float* x   = (const float*)d_in[0];
    const float* t   = (const float*)d_in[1];
    const float* w1  = (const float*)d_in[2];
    const float* b1  = (const float*)d_in[3];
    const float* w2  = (const float*)d_in[4];
    const float* b2  = (const float*)d_in[5];
    const float* mw  = (const float*)d_in[6];
    const float* mb  = (const float*)d_in[7];
    float* out = (float*)d_out;
    char* ws = (char*)d_ws;

    float*  modbuf = (float*)(ws + MOD_OFF);
    __bf16* wbf    = (__bf16*)(ws + WBF_OFF);
    float2* specA  = (float2*)(ws + SPECA_OFF);   // also MLP output (specC)
    __bf16* sbr    = (__bf16*)(ws + SBR_OFF);
    __bf16* sbi    = (__bf16*)(ws + SBI_OFF);

    k_mod  <<<dim3(BSZ * 2 * DIMC / 256), dim3(256), 0, stream>>>(t, mw, mb, modbuf);
    k_wconv<<<dim3((NB * 4 * BS * BS + 255) / 256), dim3(256), 0, stream>>>(w1, w2, wbf);
    k_rowfft<<<dim3(BSZ * DIMC * HH), dim3(128), 0, stream>>>(x, specA);
    k_colfft<<<dim3(BSZ * DIMC * WF), dim3(128), 0, stream>>>(specA, sbr, sbi);
    k_mlp  <<<dim3((unsigned)(NTILES / 4), NB), dim3(128), 0, stream>>>(
               sbr, sbi, wbf, b1, b2, modbuf, (float*)specA);
    k_icolfft<<<dim3(BSZ * DIMC * WF), dim3(128), 0, stream>>>(specA);
    k_irowfft<<<dim3(BSZ * DIMC * HH), dim3(128), 0, stream>>>(specA, x, out);
}
